// Couplformer_27084063768849
// MI455X (gfx1250) — matrix operand order repaired and checked by reading, not compiled, not run on hardware
//
#include <hip/hip_runtime.h>
#include <hip/hip_bf16.h>

#define B_   16
#define N_   4096
#define C_   384
#define NH_  12
#define HD_  32
#define HT_  64
#define WD_  64
#define SCALE_ 0.4204482076268573f   // 32^-0.25

typedef __attribute__((ext_vector_type(16))) _Float16 v16h;
typedef __attribute__((ext_vector_type(8)))  _Float16 v8h;
typedef __attribute__((ext_vector_type(8)))  float    v8f;

static __device__ __forceinline__ v8f wmma_f16(v16h a, v16h b, v8f c) {
  return __builtin_amdgcn_wmma_f32_16x16x32_f16(false, a, false, b, (short)0, c, false, false);
}

// ---- A-operand fragment (16x32 f16), K contiguous. element (m,K) at p0+m*sM+K.
// lanes 0-15: row=lane, K=0..15 ; lanes 16-31: row=lane-16, K=16..31
// => two 16-byte loads per lane.
static __device__ __forceinline__ v16h load_a_f16c(const _Float16* p0, int sM, int lane) {
  int r  = lane & 15;
  int kb = (lane >> 4) << 4;
  const _Float16* p = p0 + (size_t)r * sM + kb;
  v8h lo = *(const v8h*)(p);
  v8h hi = *(const v8h*)(p + 8);
  return __builtin_shufflevector(lo, hi, 0,1,2,3,4,5,6,7,8,9,10,11,12,13,14,15);
}

// ---- B-operand fragment (32x16 f16), Bt[n][k] row-major (K contiguous).
// lanes 0-15: col=lane, K=0..15 ; lanes 16-31: col=lane-16, K=16..31
// => two 16-byte loads per lane.
static __device__ __forceinline__ v16h load_b_f16c(const _Float16* p0, int sN, int lane) {
  int n  = lane & 15;
  int kb = (lane >> 4) << 4;
  const _Float16* p = p0 + (size_t)n * sN + kb;
  v8h b0 = *(const v8h*)(p);
  v8h b1 = *(const v8h*)(p + 8);
  return __builtin_shufflevector(b0, b1, 0,1,2,3,4,5,6,7,8,9,10,11,12,13,14,15);
}

// ---- B-operand fragment from f32 weights (K contiguous) via float4 loads ----
static __device__ __forceinline__ v16h load_b_f32c(const float* p0, int sN, int lane) {
  int n  = lane & 15;
  int kb = (lane >> 4) << 4;
  const float* p = p0 + (size_t)n * sN + kb;
  float4 f0 = *(const float4*)(p);
  float4 f1 = *(const float4*)(p + 4);
  float4 f2 = *(const float4*)(p + 8);
  float4 f3 = *(const float4*)(p + 12);
  v16h b;
  b[0]=(_Float16)f0.x;  b[1]=(_Float16)f0.y;  b[2]=(_Float16)f0.z;  b[3]=(_Float16)f0.w;
  b[4]=(_Float16)f1.x;  b[5]=(_Float16)f1.y;  b[6]=(_Float16)f1.z;  b[7]=(_Float16)f1.w;
  b[8]=(_Float16)f2.x;  b[9]=(_Float16)f2.y;  b[10]=(_Float16)f2.z; b[11]=(_Float16)f2.w;
  b[12]=(_Float16)f3.x; b[13]=(_Float16)f3.y; b[14]=(_Float16)f3.z; b[15]=(_Float16)f3.w;
  return b;
}

// ============================================================================
// Kernel 1: qkv = x @ Wqkv^T + b.  q,k -> [B,NH,N,HD] ; v -> TRANSPOSED [B,NH,HD,N]
// grid = (M/16)*9 blocks, 256 threads (8 waves). Block covers 16 rows x 128 cols.
// ============================================================================
__global__ __launch_bounds__(256) void qkv_kernel(const float* __restrict__ x,
    const float* __restrict__ Wqkv, const float* __restrict__ bqkv,
    _Float16* __restrict__ q, _Float16* __restrict__ k, _Float16* __restrict__ v) {
  __shared__ _Float16 As[16 * 32];
  const int tid = threadIdx.x, lane = tid & 31, wid = tid >> 5;
  const int band = blockIdx.x / 9, nblk = blockIdx.x % 9;
  const int m0 = band * 16;
  const int n0 = (nblk * 8 + wid) * 16;
  const int er = (tid * 2) >> 5, ec = (tid * 2) & 31;   // staging: 2 elems/thread
  v8f acc = {};
  for (int k0 = 0; k0 < C_; k0 += 32) {
    {
      const float2 xv = *(const float2*)(x + (size_t)(m0 + er) * C_ + k0 + ec);
      union { _Float16 h[2]; uint32_t u; } pk;
      pk.h[0] = (_Float16)xv.x; pk.h[1] = (_Float16)xv.y;
      *(uint32_t*)&As[tid * 2] = pk.u;
    }
    __syncthreads();
    v16h a = load_a_f16c(As, 32, lane);
    v16h b = load_b_f32c(Wqkv + (size_t)n0 * C_ + k0, C_, lane);
    acc = wmma_f16(a, b, acc);
    __syncthreads();
  }
  const int nl = lane & 15, mb = (lane >> 4) << 3;
  const int gn = n0 + nl;
  const int s = gn / C_, j = gn % C_;
  const int h = j >> 5, c = j & 31;
  const float bias = bqkv[gn];
#pragma unroll
  for (int r = 0; r < 8; ++r) {
    int gm = m0 + mb + r;
    int bq = gm >> 12, nt = gm & (N_ - 1);
    _Float16 val = (_Float16)(acc[r] + bias);
    if (s == 2) {  // v transposed: [bh][c][token]
      v[((size_t)(bq * NH_ + h) * HD_ + c) * N_ + nt] = val;
    } else {
      _Float16* dst = (s == 0) ? q : k;
      dst[(size_t)((bq * NH_ + h) * N_ + nt) * HD_ + c] = val;
    }
  }
}

// ============================================================================
// Kernel 2: attention scores + softmax.  grid=(192, 2), 512 threads (16 waves).
// mode 0 (height): S[y,z] = sum_{w,c} q[y,w,c] k[z,w,c]   (row stride 2048)
// mode 1 (width) : S[w,v] = sum_{y,c} q[y,w,c] k[y,v,c]   (row stride 32, step 2048)
// K is contiguous within each 32-step in BOTH modes -> vector fragment loads.
// ============================================================================
__global__ __launch_bounds__(512) void scores_kernel(const _Float16* __restrict__ q,
    const _Float16* __restrict__ k,
    _Float16* __restrict__ aprob, _Float16* __restrict__ bprob) {
  __shared__ float Sm[64 * 64];
  const int bh = blockIdx.x, mode = blockIdx.y;
  const int tid = threadIdx.x, lane = tid & 31, wid = tid >> 5;
  const int m0 = (wid >> 2) * 16, n0 = (wid & 3) * 16;
  const _Float16* qbh = q + (size_t)bh * N_ * HD_;
  const _Float16* kbh = k + (size_t)bh * N_ * HD_;
  const int sRow = mode ? HD_ : (WD_ * HD_);      // 32 : 2048
  const int adv  = mode ? (WD_ * HD_) : 32;       // per-K-step base advance
  const _Float16* pa = qbh + (size_t)m0 * sRow;
  const _Float16* pb = kbh + (size_t)n0 * sRow;
  v8f acc = {};
  for (int t = 0; t < 64; ++t) {                  // K = 2048 = 64 x 32
    if (t + 16 < 64) {
      __builtin_prefetch(pa + 16 * adv, 0, 1);
      __builtin_prefetch(pb + 16 * adv, 0, 1);
    }
    v16h a = load_a_f16c(pa, sRow, lane);
    v16h b = load_b_f16c(pb, sRow, lane);
    acc = wmma_f16(a, b, acc);
    pa += adv; pb += adv;
  }
  const int nl = lane & 15, mb = (lane >> 4) << 3;
#pragma unroll
  for (int r = 0; r < 8; ++r)
    Sm[(m0 + mb + r) * 64 + n0 + nl] = acc[r] * SCALE_;
  __syncthreads();
  _Float16* out = (mode ? bprob : aprob) + (size_t)bh * 64 * 64;
  if (tid < 64) {
    float mx = -1e30f;
    for (int j = 0; j < 64; ++j) mx = fmaxf(mx, Sm[tid * 64 + j]);
    float ssum = 0.f;
    for (int j = 0; j < 64; ++j) ssum += __expf(Sm[tid * 64 + j] - mx);
    const float inv = 1.f / ssum;
    for (int j = 0; j < 64; j += 2) {
      union { _Float16 h[2]; uint32_t u; } pk;
      pk.h[0] = (_Float16)(__expf(Sm[tid * 64 + j]     - mx) * inv);
      pk.h[1] = (_Float16)(__expf(Sm[tid * 64 + j + 1] - mx) * inv);
      *(uint32_t*)&out[tid * 64 + j] = pk.u;
    }
  }
}

// ============================================================================
// Kernel 3: per (b,h): for each channel c:
//   tmp[y,v] = sum_w V_c[y,w] * Bp[v,w]   (V_c contiguous rows from vT layout)
//   out[u,v] = sum_y Ap[u,y]  * tmp[y,v]  (tmp kept transposed in LDS: TmT[v][y])
// stores f16 into out_t[B, N, C] (token = u*64+v, feature = h*32+c)
// grid = 192 blocks, 512 threads (16 waves, one 16x16 tile each).
// ============================================================================
__global__ __launch_bounds__(512) void outmix_kernel(const _Float16* __restrict__ vbuf,
    const _Float16* __restrict__ aprob, const _Float16* __restrict__ bprob,
    _Float16* __restrict__ out_t) {
  __shared__ _Float16 Ap[64 * 64];
  __shared__ _Float16 Bp[64 * 64];
  __shared__ _Float16 TmT[64 * 64];   // transposed tmp: TmT[v][y]
  const int bh = blockIdx.x;
  const int tid = threadIdx.x, lane = tid & 31, wid = tid >> 5;
  const int bq = bh / NH_, h = bh % NH_;
  {
    const uint32_t* ap32 = (const uint32_t*)(aprob + (size_t)bh * 4096);
    const uint32_t* bp32 = (const uint32_t*)(bprob + (size_t)bh * 4096);
    uint32_t* Ap32 = (uint32_t*)Ap;
    uint32_t* Bp32 = (uint32_t*)Bp;
#pragma unroll
    for (int t = 0; t < 4; ++t) { Ap32[tid + t * 512] = ap32[tid + t * 512];
                                  Bp32[tid + t * 512] = bp32[tid + t * 512]; }
  }
  __syncthreads();
  const _Float16* vbh = vbuf + (size_t)bh * HD_ * N_;   // vT: [c][token]
  const int m0 = (wid >> 2) * 16, n0 = (wid & 3) * 16;
  const int nl = lane & 15, mb = (lane >> 4) << 3;
  for (int c = 0; c < HD_; ++c) {
    // GEMM1: tmp = V_c @ Bp^T   (A elem (y,w) at c*4096 + y*64 + w: contiguous)
    v8f acc = {};
#pragma unroll
    for (int t = 0; t < 2; ++t) {
      v16h a = load_a_f16c(vbh + (size_t)c * N_ + m0 * 64 + t * 32, 64, lane);
      v16h b = load_b_f16c(Bp + n0 * 64 + t * 32, 64, lane);   // Bt[v][w]
      acc = wmma_f16(a, b, acc);
    }
    {   // store transposed: TmT[v][y] ; 8 contiguous halves per lane
      v8h t8;
#pragma unroll
      for (int r = 0; r < 8; ++r) t8[r] = (_Float16)acc[r];
      *(v8h*)&TmT[(n0 + nl) * 64 + m0 + mb] = t8;
    }
    __syncthreads();
    // GEMM2: out = Ap @ tmp   (B operand = TmT[n=v][k=y], contiguous)
    v8f acc2 = {};
#pragma unroll
    for (int t = 0; t < 2; ++t) {
      v16h a = load_a_f16c(Ap + m0 * 64 + t * 32, 64, lane);
      v16h b = load_b_f16c(TmT + n0 * 64 + t * 32, 64, lane);
      acc2 = wmma_f16(a, b, acc2);
    }
#pragma unroll
    for (int r = 0; r < 8; ++r) {
      int u = m0 + mb + r, vv = n0 + nl;
      out_t[((size_t)bq * N_ + u * WD_ + vv) * C_ + h * HD_ + c] = (_Float16)acc2[r];
    }
    __syncthreads();  // before TmT is overwritten for next c
  }
}

// ============================================================================
// Kernel 4: final projection  out = out_t(f16) @ Wp^T + bp  (fp32 out)
// grid = (M/16)*3 blocks, 256 threads (8 waves). Block covers 16 x 128.
// ============================================================================
__global__ __launch_bounds__(256) void proj_kernel(const _Float16* __restrict__ xt,
    const float* __restrict__ Wp, const float* __restrict__ bp,
    float* __restrict__ out) {
  __shared__ _Float16 As[16 * 32];
  const int tid = threadIdx.x, lane = tid & 31, wid = tid >> 5;
  const int band = blockIdx.x / 3, nblk = blockIdx.x % 3;
  const int m0 = band * 16;
  const int n0 = (nblk * 8 + wid) * 16;
  const int er = (tid * 2) >> 5, ec = (tid * 2) & 31;
  v8f acc = {};
  for (int k0 = 0; k0 < C_; k0 += 32) {
    *(uint32_t*)&As[tid * 2] =
        *(const uint32_t*)(xt + (size_t)(m0 + er) * C_ + k0 + ec);
    __syncthreads();
    v16h a = load_a_f16c(As, 32, lane);
    v16h b = load_b_f32c(Wp + (size_t)n0 * C_ + k0, C_, lane);
    acc = wmma_f16(a, b, acc);
    __syncthreads();
  }
  const int nl = lane & 15, mb = (lane >> 4) << 3;
  const int gn = n0 + nl;
  const float bias = bp[gn];
#pragma unroll
  for (int r = 0; r < 8; ++r) {
    int gm = m0 + mb + r;
    out[(size_t)gm * C_ + gn] = acc[r] + bias;
  }
}

extern "C" void kernel_launch(void* const* d_in, const int* in_sizes, int n_in,
                              void* d_out, int out_size, void* d_ws, size_t ws_size,
                              hipStream_t stream) {
  (void)in_sizes; (void)n_in; (void)out_size; (void)ws_size;
  const float* x    = (const float*)d_in[0];
  const float* Wqkv = (const float*)d_in[1];
  const float* bqkv = (const float*)d_in[2];
  const float* Wp   = (const float*)d_in[3];
  const float* bp   = (const float*)d_in[4];
  float* out = (float*)d_out;

  char* ws = (char*)d_ws;
  const size_t QS = (size_t)B_ * NH_ * N_ * HD_ * sizeof(_Float16);   // 50.3 MB each
  const size_t PS = (size_t)B_ * NH_ * 64 * 64 * sizeof(_Float16);    // 1.57 MB each
  _Float16* q   = (_Float16*)(ws);
  _Float16* k   = (_Float16*)(ws + QS);
  _Float16* v   = (_Float16*)(ws + 2 * QS);
  _Float16* ap  = (_Float16*)(ws + 3 * QS);
  _Float16* bpr = (_Float16*)(ws + 3 * QS + PS);
  _Float16* ot  = (_Float16*)(ws + 3 * QS + 2 * PS);

  qkv_kernel  <<<dim3((B_ * N_ / 16) * 9), dim3(256), 0, stream>>>(x, Wqkv, bqkv, q, k, v);
  scores_kernel<<<dim3(B_ * NH_, 2),       dim3(512), 0, stream>>>(q, k, ap, bpr);
  outmix_kernel<<<dim3(B_ * NH_),          dim3(512), 0, stream>>>(v, ap, bpr, ot);
  proj_kernel <<<dim3((B_ * N_ / 16) * 3), dim3(256), 0, stream>>>(ot, Wp, bp, out);
}
